// MultiHeadAttention_64750926954850
// MI455X (gfx1250) — matrix operand order repaired and checked by reading, not compiled, not run on hardware
//
#include <hip/hip_runtime.h>
#include <hip/hip_bf16.h>

// ---------------------------------------------------------------------------
// MHA forward for MI455X (gfx1250, wave32, WMMA, async-LDS staging).
//   Prep   : fp32 -> bf16 conversions (x; W_qkv^T; W_out^T) so all GEMM tile
//            staging is a raw byte copy (async-to-LDS capable).
//   Stage 1: QKV GEMM (bf16 WMMA), epilogue scatters Q(scaled)/K row-major and
//            V transposed [B,H,d,T].
//   Stage 2: causal flash attention, double-buffered K/V tiles, full-wave
//            online softmax, WMMA for S=QK^T and O+=P*V.
//   Stage 3: output projection GEMM -> fp32 + bias.
// ---------------------------------------------------------------------------

#ifndef __has_builtin
#define __has_builtin(x) 0
#endif

#if __has_builtin(__builtin_amdgcn_global_load_async_to_lds_b128) && \
    __has_builtin(__builtin_amdgcn_s_wait_asynccnt)
#define USE_ASYNC 1
#else
#define USE_ASYNC 0
#endif

typedef __attribute__((ext_vector_type(16))) __bf16 v16bf;
typedef __attribute__((ext_vector_type(8)))  __bf16 v8bf;
typedef __attribute__((ext_vector_type(8)))  float  v8f;
typedef __attribute__((ext_vector_type(4)))  int    v4i;

#define NHEADS 12
#define TSEQ   4096
#define DMODEL 768
#define N3     2304
#define DK     64

static __device__ __forceinline__ v8f vzero8() {
  v8f z = {0.f, 0.f, 0.f, 0.f, 0.f, 0.f, 0.f, 0.f};
  return z;
}

// Copy 16 bytes (8 bf16) global -> LDS.  Async DMA on gfx1250 toolchains that
// expose the builtin (param type: v4i in AS1/AS3); plain vector copy otherwise.
static __device__ __forceinline__ void copy16B(const __bf16* g, __bf16* l) {
#if USE_ASYNC
  __builtin_amdgcn_global_load_async_to_lds_b128(
      (__attribute__((address_space(1))) v4i*)(uintptr_t)g,
      (__attribute__((address_space(3))) v4i*)(uintptr_t)l, 0, 0);
#else
  *(v8bf*)l = *(const v8bf*)g;
#endif
}

static __device__ __forceinline__ void stage_wait() {
#if USE_ASYNC
  __builtin_amdgcn_s_wait_asynccnt(0);
#endif
  __syncthreads();
}

// A fragment, 16x32 bf16 (ISA 7.12.2): lane l holds row m=l%16;
// elems 0..7 = K [kb..kb+7], elems 8..15 = K [16+kb..16+kb+7], kb=(l/16)*8.
static __device__ __forceinline__ v16bf load_a_frag(const __bf16* smem, int rowBase,
                                                    int rowStride, int k0, int lane) {
  int m  = rowBase + (lane & 15);
  int kb = (lane >> 4) * 8;
  const __bf16* p = smem + m * rowStride + k0;
  v8bf lo = *(const v8bf*)(p + kb);
  v8bf hi = *(const v8bf*)(p + 16 + kb);
  return __builtin_shufflevector(lo, hi, 0, 1, 2, 3, 4, 5, 6, 7,
                                 8, 9, 10, 11, 12, 13, 14, 15);
}

// B fragment, 32x16 bf16 from an n-major (transposed) LDS image Bt[n][k]:
// lane l holds col n=l%16; elems 0..7 = K [kb..kb+7], elems 8..15 = K [16+kb..16+kb+7], kb=k0+(l/16)*8.
static __device__ __forceinline__ v16bf load_b_frag(const __bf16* smem, int nBase,
                                                    int rowStride, int k0, int lane) {
  int n  = nBase + (lane & 15);
  int kb = k0 + (lane >> 4) * 8;
  const __bf16* p = smem + n * rowStride + kb;
  v8bf lo = *(const v8bf*)(p);
  v8bf hi = *(const v8bf*)(p + 16);
  return __builtin_shufflevector(lo, hi, 0, 1, 2, 3, 4, 5, 6, 7,
                                 8, 9, 10, 11, 12, 13, 14, 15);
}

static __device__ __forceinline__ v8f wmma_bf16(v16bf a, v16bf b, v8f c) {
  return __builtin_amdgcn_wmma_f32_16x16x32_bf16(false, a, false, b,
                                                 (short)0, c, false, false);
}

// ---------------------------------------------------------------------------
// Prep kernels: fp32 -> bf16 (plain and transposing).
// ---------------------------------------------------------------------------
__global__ __launch_bounds__(256) void cvt_kernel(const float* __restrict__ in,
                                                  __bf16* __restrict__ out, int n) {
  int i = (blockIdx.x * 256 + threadIdx.x) * 8;
  if (i < n) {
    float4 a = *(const float4*)(in + i);
    float4 b = *(const float4*)(in + i + 4);
    v8bf o;
    o[0] = (__bf16)a.x; o[1] = (__bf16)a.y; o[2] = (__bf16)a.z; o[3] = (__bf16)a.w;
    o[4] = (__bf16)b.x; o[5] = (__bf16)b.y; o[6] = (__bf16)b.z; o[7] = (__bf16)b.w;
    *(v8bf*)(out + i) = o;
  }
}

// in[K][N] f32 (row-major) -> out[N][K] bf16 (n-major).
__global__ __launch_bounds__(256) void cvtT_kernel(const float* __restrict__ in,
                                                   __bf16* __restrict__ out,
                                                   int K, int N) {
  int idx = blockIdx.x * 256 + threadIdx.x;
  if (idx < K * N) {
    int k = idx / N, n = idx % N;
    out[(size_t)n * K + k] = (__bf16)in[idx];
  }
}

// Stage a 64x32 bf16 tile (row-major, stride `gstride`) into LDS [64][32].
static __device__ __forceinline__ void stage_tile64x32(const __bf16* gbase,
                                                       size_t gstride,
                                                       __bf16* sbuf, int tid) {
  int row = tid >> 1, kh = (tid & 1) * 16;
  const __bf16* gp = gbase + (size_t)row * gstride + kh;
  __bf16*       sp = sbuf + row * 32 + kh;
  copy16B(gp, sp);
  copy16B(gp + 8, sp + 8);
}

// ---------------------------------------------------------------------------
// Stage 1: QKV GEMM.  Xb[8192,768] (bf16) x Wt[2304][768] (bf16, n-major) + b.
// Tile 64x64, K-step 32, double-buffered, 4 waves.
// ---------------------------------------------------------------------------
__global__ __launch_bounds__(128) void qkv_gemm_kernel(
    const __bf16* __restrict__ Xb, const __bf16* __restrict__ Wt,
    const float* __restrict__ bias,
    __bf16* __restrict__ Q, __bf16* __restrict__ K, __bf16* __restrict__ Vt) {
  __shared__ __attribute__((aligned(64))) __bf16 As[2][64 * 32];
  __shared__ __attribute__((aligned(64))) __bf16 Bts[2][64 * 32];

  const int tid  = threadIdx.x;
  const int wave = tid >> 5;
  const int lane = tid & 31;
  const int m0   = blockIdx.x * 64;
  const int n0   = blockIdx.y * 64;

  v8f acc[4];
#pragma unroll
  for (int f = 0; f < 4; ++f) acc[f] = vzero8();

  stage_tile64x32(Xb + (size_t)m0 * DMODEL, DMODEL, As[0], tid);
  stage_tile64x32(Wt + (size_t)n0 * DMODEL, DMODEL, Bts[0], tid);
  stage_wait();

  for (int ks = 0; ks < DMODEL / 32; ++ks) {
    const int cur = ks & 1;
    if (ks + 1 < DMODEL / 32) {
      stage_tile64x32(Xb + (size_t)m0 * DMODEL + (ks + 1) * 32, DMODEL, As[cur ^ 1], tid);
      stage_tile64x32(Wt + (size_t)n0 * DMODEL + (ks + 1) * 32, DMODEL, Bts[cur ^ 1], tid);
    }
    v16bf a = load_a_frag(As[cur], wave * 16, 32, 0, lane);
    v16bf bfr[4];
#pragma unroll
    for (int f = 0; f < 4; ++f) bfr[f] = load_b_frag(Bts[cur], f * 16, 32, 0, lane);
#pragma unroll
    for (int f = 0; f < 4; ++f) acc[f] = wmma_bf16(a, bfr[f], acc[f]);
    stage_wait();
  }

  // Epilogue: col -> (head, q/k/v, d).  Q gets 1/sqrt(64) folded in.
  // Q,K -> [B,H,T,64]; V -> transposed [B,H,64,T].
  const int mBase = m0 + wave * 16;
#pragma unroll
  for (int f = 0; f < 4; ++f) {
    int   col = n0 + f * 16 + (lane & 15);
    int   h   = col / 192;
    int   s   = (col % 192) / 64;
    int   d   = col % 64;
    float bv  = bias[col];
#pragma unroll
    for (int i = 0; i < 8; ++i) {
      int    m   = mBase + i + ((lane >> 4) << 3);
      int    bb  = m >> 12;
      int    t   = m & 4095;
      float  val = acc[f][i] + bv;
      if (s == 0) {
        Q[(((size_t)bb * NHEADS + h) * TSEQ + t) * DK + d] = (__bf16)(val * 0.125f);
      } else if (s == 1) {
        K[(((size_t)bb * NHEADS + h) * TSEQ + t) * DK + d] = (__bf16)val;
      } else {
        Vt[(((size_t)bb * NHEADS + h) * DK + d) * TSEQ + t] = (__bf16)val;
      }
    }
  }
}

// ---------------------------------------------------------------------------
// Stage 2: causal flash attention.  One block per (qtile=64, b*h).
// Double-buffered K/V^T tiles, full-wave online softmax.
// ---------------------------------------------------------------------------
__global__ __launch_bounds__(128) void attn_kernel(
    const __bf16* __restrict__ Q, const __bf16* __restrict__ K,
    const __bf16* __restrict__ Vt, __bf16* __restrict__ ctx) {
  __shared__ __attribute__((aligned(64))) __bf16 Qs[64 * 64];      // [q][d]
  __shared__ __attribute__((aligned(64))) __bf16 Ks[2][64 * 64];   // [key][d]
  __shared__ __attribute__((aligned(64))) __bf16 Vts[2][64 * 64];  // [d][key]
  __shared__ __attribute__((aligned(64))) float  Ss[64 * 64];      // wave-private strips
  __shared__ __attribute__((aligned(64))) __bf16 Ps[64 * 64];      // wave-private strips

  const int    tid   = threadIdx.x;
  const int    wave  = tid >> 5;
  const int    lane  = tid & 31;
  const int    qt    = blockIdx.x;
  const int    bh    = blockIdx.y;
  const size_t base  = (size_t)bh * TSEQ * DK;   // Q/K row-major base
  const size_t basev = (size_t)bh * DK * TSEQ;   // V^T base

  const int row = tid >> 1;          // staging row 0..63
  const int kh  = (tid & 1) * 32;    // staging 32-element half

  // Stage Q tile once (row-major copy).
  {
    const __bf16* gp = Q + base + (size_t)(qt * 64 + row) * DK + kh;
#pragma unroll
    for (int j = 0; j < 32; j += 8) copy16B(gp + j, Qs + row * 64 + kh + j);
  }
  // Stage K/V^T tile 0 into buffer 0.
  {
    const __bf16* gk = K + base + (size_t)row * DK + kh;
    const __bf16* gv = Vt + basev + (size_t)row * TSEQ + kh;
#pragma unroll
    for (int j = 0; j < 32; j += 8) {
      copy16B(gk + j, Ks[0] + row * 64 + kh + j);
      copy16B(gv + j, Vts[0] + row * 64 + kh + j);
    }
  }
  stage_wait();

  const v16bf qa0 = load_a_frag(Qs, wave * 16, 64, 0, lane);
  const v16bf qa1 = load_a_frag(Qs, wave * 16, 64, 32, lane);

  v8f o[4];
#pragma unroll
  for (int f = 0; f < 4; ++f) o[f] = vzero8();
  float mrow = -INFINITY;   // replicated in lane r and r+16
  float lrow = 0.f;

  for (int kt = 0; kt <= qt; ++kt) {
    const int cur = kt & 1;
    if (kt < qt) {  // prefetch next key tile into the other buffer
      const __bf16* gk = K + base + (size_t)((kt + 1) * 64 + row) * DK + kh;
      const __bf16* gv = Vt + basev + (size_t)row * TSEQ + (kt + 1) * 64 + kh;
#pragma unroll
      for (int j = 0; j < 32; j += 8) {
        copy16B(gk + j, Ks[cur ^ 1] + row * 64 + kh + j);
        copy16B(gv + j, Vts[cur ^ 1] + row * 64 + kh + j);
      }
    }

    // S = Q K^T for this wave's 16 rows x 64 keys (K-dim = d = 64 -> 2 WMMA).
    v8f s[4];
#pragma unroll
    for (int f = 0; f < 4; ++f) {
      v16bf b0 = load_b_frag(Ks[cur], f * 16, 64, 0, lane);
      v16bf b1 = load_b_frag(Ks[cur], f * 16, 64, 32, lane);
      v8f   z  = vzero8();
      z        = wmma_bf16(qa0, b0, z);
      z        = wmma_bf16(qa1, b1, z);
      s[f]     = z;
    }

    // Spill S to the wave-private LDS strip (C layout: row = i + 8*(lane/16)).
    float* sw = Ss + wave * 16 * 64;
#pragma unroll
    for (int f = 0; f < 4; ++f)
#pragma unroll
      for (int i = 0; i < 8; ++i)
        sw[(i + ((lane >> 4) << 3)) * 64 + f * 16 + (lane & 15)] = s[f][i];

    // Online softmax, all 32 lanes: lane r / r+16 own halves of query row r.
    float alpha;
    {
      int    r    = lane & 15;
      int    ch   = (lane >> 4) * 32;
      int    qg   = qt * 64 + wave * 16 + r;
      int    clim = qg - kt * 64;          // columns c <= clim are unmasked
      float* rp   = sw + r * 64 + ch;
      float  mx   = -3.0e38f;
#pragma unroll 8
      for (int c = 0; c < 32; ++c) {
        float v = ((ch + c) <= clim) ? rp[c] : -3.0e38f;  // scale pre-folded
        rp[c]   = v;
        mx      = fmaxf(mx, v);
      }
      mx         = fmaxf(mx, __shfl_xor(mx, 16, 32));
      float mnew = fmaxf(mrow, mx);
      alpha      = __expf(mrow - mnew);
      float   sum  = 0.f;
      __bf16* prow = Ps + (wave * 16 + r) * 64 + ch;
#pragma unroll 8
      for (int c = 0; c < 32; ++c) {
        float p = __expf(rp[c] - mnew);
        sum += p;
        prow[c] = (__bf16)p;
      }
      sum  = sum + __shfl_xor(sum, 16, 32);
      lrow = lrow * alpha + sum;
      mrow = mnew;
    }

    // Rescale O by per-row alpha (row r lives in lane r).
#pragma unroll
    for (int i = 0; i < 8; ++i) {
      float av = __shfl(alpha, i + ((lane >> 4) << 3), 32);
#pragma unroll
      for (int f = 0; f < 4; ++f) o[f][i] *= av;
    }

    // O += P V  (K-dim = keys = 64 -> 2 WMMA per d-fragment).
    v16bf pa0 = load_a_frag(Ps, wave * 16, 64, 0, lane);
    v16bf pa1 = load_a_frag(Ps, wave * 16, 64, 32, lane);
#pragma unroll
    for (int f = 0; f < 4; ++f) {
      v16bf vb0 = load_b_frag(Vts[cur], f * 16, 64, 0, lane);
      v16bf vb1 = load_b_frag(Vts[cur], f * 16, 64, 32, lane);
      o[f]      = wmma_bf16(pa0, vb0, o[f]);
      o[f]      = wmma_bf16(pa1, vb1, o[f]);
    }
    stage_wait();   // next tile landed; buffers safe to swap
  }

  // Final 1/l normalization and bf16 context store [B*T, 768].
  const float linv = 1.0f / lrow;            // valid in all lanes (replicated)
  const int   b = bh / NHEADS, h = bh % NHEADS;
#pragma unroll
  for (int i = 0; i < 8; ++i) {
    float  lv = __shfl(linv, i + ((lane >> 4) << 3), 32);
    int    tq = qt * 64 + wave * 16 + i + ((lane >> 4) << 3);
    size_t ro = ((size_t)b * TSEQ + tq) * DMODEL + h * DK;
#pragma unroll
    for (int f = 0; f < 4; ++f) {
      int d = f * 16 + (lane & 15);
      ctx[ro + d] = (__bf16)(o[f][i] * lv);
    }
  }
}

// ---------------------------------------------------------------------------
// Stage 3: output projection.  ctx[8192,768] (bf16) x Woutt[768][768]
// (bf16, n-major) + b_out -> fp32.
// ---------------------------------------------------------------------------
__global__ __launch_bounds__(128) void out_gemm_kernel(
    const __bf16* __restrict__ A, const __bf16* __restrict__ Wt,
    const float* __restrict__ bias, float* __restrict__ out) {
  __shared__ __attribute__((aligned(64))) __bf16 As[2][64 * 32];
  __shared__ __attribute__((aligned(64))) __bf16 Bts[2][64 * 32];

  const int tid  = threadIdx.x;
  const int wave = tid >> 5;
  const int lane = tid & 31;
  const int m0   = blockIdx.x * 64;
  const int n0   = blockIdx.y * 64;

  v8f acc[4];
#pragma unroll
  for (int f = 0; f < 4; ++f) acc[f] = vzero8();

  stage_tile64x32(A + (size_t)m0 * DMODEL, DMODEL, As[0], tid);
  stage_tile64x32(Wt + (size_t)n0 * DMODEL, DMODEL, Bts[0], tid);
  stage_wait();

  for (int ks = 0; ks < DMODEL / 32; ++ks) {
    const int cur = ks & 1;
    if (ks + 1 < DMODEL / 32) {
      stage_tile64x32(A + (size_t)m0 * DMODEL + (ks + 1) * 32, DMODEL, As[cur ^ 1], tid);
      stage_tile64x32(Wt + (size_t)n0 * DMODEL + (ks + 1) * 32, DMODEL, Bts[cur ^ 1], tid);
    }
    v16bf a = load_a_frag(As[cur], wave * 16, 32, 0, lane);
    v16bf bfr[4];
#pragma unroll
    for (int f = 0; f < 4; ++f) bfr[f] = load_b_frag(Bts[cur], f * 16, 32, 0, lane);
#pragma unroll
    for (int f = 0; f < 4; ++f) acc[f] = wmma_bf16(a, bfr[f], acc[f]);
    stage_wait();
  }

  const int mBase = m0 + wave * 16;
#pragma unroll
  for (int f = 0; f < 4; ++f) {
    int   col = n0 + f * 16 + (lane & 15);
    float bv  = bias[col];
#pragma unroll
    for (int i = 0; i < 8; ++i) {
      int m = mBase + i + ((lane >> 4) << 3);
      out[(size_t)m * DMODEL + col] = acc[f][i] + bv;
    }
  }
}

// ---------------------------------------------------------------------------
extern "C" void kernel_launch(void* const* d_in, const int* in_sizes, int n_in,
                              void* d_out, int out_size, void* d_ws, size_t ws_size,
                              hipStream_t stream) {
  (void)in_sizes; (void)n_in; (void)out_size; (void)ws_size;
  const float* x    = (const float*)d_in[0];
  // d_in[1] is the boolean causal mask; causality is computed analytically.
  const float* Wqkv = (const float*)d_in[2];
  const float* bqkv = (const float*)d_in[3];
  const float* Wout = (const float*)d_in[4];
  const float* bout = (const float*)d_in[5];
  float*       out  = (float*)d_out;

  // Workspace (bf16):
  //   Xb[8192*768]  Wqkvt[2304*768]  Woutt[768*768]
  //   Q,K [B,H,T,64]  Vt [B,H,64,T]  ctx [8192*768]        (~68 MB total)
  const size_t XB   = (size_t)2 * TSEQ * DMODEL;            // 6,291,456
  const size_t QKVE = (size_t)2 * NHEADS * TSEQ * DK;       // 6,291,456
  __bf16* Xb  = (__bf16*)d_ws;
  __bf16* Wqt = Xb + XB;
  __bf16* Wot = Wqt + (size_t)DMODEL * N3;
  __bf16* Qw  = Wot + (size_t)DMODEL * DMODEL;
  __bf16* Kw  = Qw + QKVE;
  __bf16* Vtw = Kw + QKVE;
  __bf16* Cw  = Vtw + QKVE;

  cvt_kernel<<<(int)(XB / (256 * 8)), 256, 0, stream>>>(x, Xb, (int)XB);
  cvtT_kernel<<<(DMODEL * N3 + 255) / 256, 256, 0, stream>>>(Wqkv, Wqt, DMODEL, N3);
  cvtT_kernel<<<(DMODEL * DMODEL + 255) / 256, 256, 0, stream>>>(Wout, Wot, DMODEL, DMODEL);

  qkv_gemm_kernel<<<dim3(128, 36), 128, 0, stream>>>(Xb, Wqt, bqkv, Qw, Kw, Vtw);
  attn_kernel<<<dim3(64, 24), 128, 0, stream>>>(Qw, Kw, Vtw, Cw);
  out_gemm_kernel<<<dim3(128, 12), 128, 0, stream>>>(Cw, Wot, bout, out);
}
